// E3NN_PhaseNet_NequIP_54692113547904
// MI455X (gfx1250) — compile-verified, matrix operand order checked by reading, not run on hardware
//
#include <hip/hip_runtime.h>
#include <hip/hip_bf16.h>

// ---------------------------------------------------------------------------
// E3NN / NequIP forward for MI455X (gfx1250, wave32, WMMA).
//
// Pipeline (all on `stream`):
//   memset(acc1/acc2/acc3)
//   k_node_init            : s0 = emb[species], sl1 = s0 * c1.lin1_Ws
//   conv1_edge (WMMA MLP)  : scatter 4 floats/edge -> acc1[N][4]
//   k_node1_post           : gate -> sfeat[N][8], vfeat[N][8][3]
//   k_node_lin1 (c2)       : sl[N][8], vl[N][8][3]
//   conv2_edge (WMMA MLP)  : scatter 112 floats/edge -> acc2[N][112]
//   k_node2_post           : gate -> sfeat, vfeat
//   k_node_lin1 (c3)       : sl, vl
//   conv3_edge (WMMA MLP)  : scatter 16 floats/edge -> acc3[N][16]
//   k_node3_head           : conv3 post + 64x64 head MLP -> d_out[N][4]
//
// All fan-in / irrep normalization constants that act per-column of the
// radial-MLP weights are folded into the f16 LDS weight staging.
// ---------------------------------------------------------------------------

typedef __attribute__((ext_vector_type(16))) _Float16 v16h;
typedef __attribute__((ext_vector_type(8)))  _Float16 v8h;
typedef __attribute__((ext_vector_type(8)))  float    v8f;

#define DEVFN __device__ __forceinline__

constexpr int   kN = 50000;
constexpr int   kE = 1600000;

constexpr float kStep     = 3.15f / 15.0f;
constexpr float kBasisAmp = 1.14136f * 7.3890560989306495f * 4.0f; // BASIS_NORM * sqrt(16)
constexpr float kISN      = 0.28867513459481287f;   // 1/sqrt(12)
constexpr float kSQ3      = 1.7320508075688772f;
constexpr float kInvSQ3   = 0.5773502691896258f;
constexpr float kInvSQ2   = 0.7071067811865475f;
constexpr float kInvSQ53  = 0.7745966692414834f;    // 1/sqrt(5/3)
constexpr float kC2       = 2.7386127875258306f;    // sqrt(15/2)
constexpr float kInv8     = 0.35355339059327373f;   // 1/sqrt(8)
constexpr float kInv16    = 0.25f;
constexpr float kInv32    = 0.17677669529663687f;   // 1/sqrt(32)
constexpr float kMlpO     = 0.125f;                 // 1/sqrt(64) (2nd MLP layer fan-in)

struct ColScales { float s[6]; };                   // per 8-column group of W2

DEVFN float rcpf(float x)  { return __builtin_amdgcn_rcpf(x); }       // v_rcp_f32
DEVFN float siluf(float x) { return x * rcpf(1.0f + __expf(-x)); }
DEVFN float sigmf(float x) { return rcpf(1.0f + __expf(-x)); }

DEVFN v8f wmma_f16(v16h a, v16h b, v8f c) {
  // D(f32 16x16) = A(f16 16x32) * B(f16 32x16) + C
  return __builtin_amdgcn_wmma_f32_16x16x32_f16(
      /*neg_a=*/false, a, /*neg_b=*/false, b,
      /*c_mod=*/(short)0, c, /*reuse_a=*/false, /*reuse_b=*/false);
}

// Fragment halves live in two contiguous 8-half (16B) runs -> 2x ds_load_b128.
DEVFN v16h ld_frag(const _Float16* p0, const _Float16* p1) {
  const v8h lo = *(const v8h*)p0;
  const v8h hi = *(const v8h*)p1;
  v16h r;
#pragma unroll
  for (int j = 0; j < 8; ++j) { r[j] = lo[j]; r[j + 8] = hi[j]; }
  return r;
}

// ---------------------------------------------------------------------------
// Shared radial-MLP stage: per-block edge basis -> WMMA(hb@(0.25*W1)) -> silu
// -> WMMA(H@(scale*W2)) -> sOut[EPB][W2C] (f32).  Weights staged TRANSPOSED
// ([N][K]) in LDS so B fragments are two b128 loads; fan-in / irrep constants
// folded into the staging.  EPB = 128 or 256; block = 256 threads / 8 waves.
// ---------------------------------------------------------------------------
template<int EPB, int W2C>
DEVFN void radial_stage(int eBase,
                        const float* __restrict__ evec,
                        const float* __restrict__ W1g,    // [16][64]
                        const float* __restrict__ W2g,    // [64][w2RealCols]
                        int w2RealCols, ColScales cs,
                        _Float16* sHB,                    // [EPB][32]   (K pad 16->32)
                        _Float16* sW1t,                   // [64][32]    W1^T, K pad
                        _Float16* sW2t,                   // [W2C][64]   W2^T, N pad
                        _Float16* sH,                     // [8][16][64] per-wave staging
                        float*    sOut,                   // [EPB][W2C]
                        float*    sN)                     // [EPB][3] unit vectors
{
  const int tid = threadIdx.x;

  // ---- cooperative transposed f16 weight staging (zero padded, scaled) ----
  for (int i = tid; i < 64 * 32; i += 256) {
    const int n = i >> 5, k = i & 31;
    sW1t[i] = (k < 16) ? (_Float16)(W1g[k * 64 + n] * 0.25f) : (_Float16)0.0f;
  }
  for (int i = tid; i < W2C * 64; i += 256) {
    const int n = i >> 6, k = i & 63;
    sW2t[i] = (n < w2RealCols) ? (_Float16)(W2g[k * w2RealCols + n] * cs.s[n >> 3])
                               : (_Float16)0.0f;
  }

  // ---- stage 1: geometry + smooth_finite basis ----
  // exponent identity: u=1+d, w=1-d  =>  -1/u - 1/w = -2/(1-d^2)
  if (EPB == 256) {
    const int el = tid;
    const int e3 = (eBase + el) * 3;
    const float vx = evec[e3 + 0], vy = evec[e3 + 1], vz = evec[e3 + 2];
    const float r = sqrtf(vx * vx + vy * vy + vz * vz);
    const float inv = rcpf(r + 1e-12f);
    sN[el * 3 + 0] = vx * inv; sN[el * 3 + 1] = vy * inv; sN[el * 3 + 2] = vz * inv;
    const float q = r * (1.0f / kStep);
    _Float16 hb[16];
#pragma unroll
    for (int i = 0; i < 16; ++i) {
      const float d = q - (float)i;
      const float t = 1.0f - d * d;
      float f = 0.0f;
      if (t > 0.0f) f = kBasisAmp * __expf(-2.0f * rcpf(t));
      hb[i] = (_Float16)f;
    }
    v8h q0, q1, z;
#pragma unroll
    for (int j = 0; j < 8; ++j) { q0[j] = hb[j]; q1[j] = hb[8 + j]; z[j] = (_Float16)0.0f; }
    v8h* row = (v8h*)(sHB + el * 32);
    row[0] = q0; row[1] = q1; row[2] = z; row[3] = z;
  } else {  // EPB == 128: two threads per edge, 8 basis values each
    const int el = tid & 127, half = tid >> 7;
    const int e3 = (eBase + el) * 3;
    const float vx = evec[e3 + 0], vy = evec[e3 + 1], vz = evec[e3 + 2];
    const float r = sqrtf(vx * vx + vy * vy + vz * vz);
    if (half == 0) {
      const float inv = rcpf(r + 1e-12f);
      sN[el * 3 + 0] = vx * inv; sN[el * 3 + 1] = vy * inv; sN[el * 3 + 2] = vz * inv;
    }
    const float q = r * (1.0f / kStep);
    v8h qv, z;
#pragma unroll
    for (int i = 0; i < 8; ++i) {
      const float d = q - (float)(half * 8 + i);
      const float t = 1.0f - d * d;
      float f = 0.0f;
      if (t > 0.0f) f = kBasisAmp * __expf(-2.0f * rcpf(t));
      qv[i] = (_Float16)f;
      z[i] = (_Float16)0.0f;
    }
    v8h* row = (v8h*)(sHB + el * 32);
    row[half] = qv; row[2 + half] = z;
  }
  __syncthreads();

  // ---- stage 2: per-wave WMMA chain ----
  const int wave = tid >> 5, lane = tid & 31;
  const int ln = lane & 15;              // A-row / B-col / C-col for this lane
  const int kb = (lane >> 4) << 3;       // K group base
  const int mb = kb;                     // C/D row base
  constexpr int TILES = EPB / 16;
  constexpr int TPW = TILES / 8;
  constexpr int NT = W2C / 16;
  _Float16* Hs = sH + wave * (16 * 64);

  // hoisted B fragments (shared across this wave's M tiles)
  v16h bW1[4];
#pragma unroll
  for (int ct = 0; ct < 4; ++ct) {
    const _Float16* p = sW1t + (ct * 16 + ln) * 32 + kb;
    bW1[ct] = ld_frag(p, p + 16);
  }
  v16h bW2[NT][2];
#pragma unroll
  for (int nt = 0; nt < NT; ++nt) {
    const _Float16* p = sW2t + (nt * 16 + ln) * 64 + kb;
    bW2[nt][0] = ld_frag(p, p + 16);
    bW2[nt][1] = ld_frag(p + 32, p + 48);
  }

#pragma unroll
  for (int t = 0; t < TPW; ++t) {
    const int tile = wave * TPW + t;
    const _Float16* At = sHB + tile * (16 * 32) + ln * 32 + kb;
    const v16h a = ld_frag(At, At + 16);

    // GEMM1: [16x32] x [32x64] (fan-in 1/4 folded into W1); silu -> f16 staging
#pragma unroll
    for (int ct = 0; ct < 4; ++ct) {
      v8f c = {};
      c = wmma_f16(a, bW1[ct], c);
#pragma unroll
      for (int r = 0; r < 8; ++r)
        Hs[(mb + r) * 64 + ct * 16 + ln] = (_Float16)siluf(c[r]);
    }
    // GEMM2: [16x64] x [64xW2C] (2 K tiles; LDS ds ops are in-order per wave)
    const _Float16* Ap = Hs + ln * 64 + kb;
    const v16h a0 = ld_frag(Ap, Ap + 16);
    const v16h a1 = ld_frag(Ap + 32, Ap + 48);
#pragma unroll
    for (int nt = 0; nt < NT; ++nt) {
      v8f c = {};
      c = wmma_f16(a0, bW2[nt][0], c);
      c = wmma_f16(a1, bW2[nt][1], c);
#pragma unroll
      for (int r = 0; r < 8; ++r)
        sOut[(tile * 16 + mb + r) * W2C + nt * 16 + ln] = c[r];
    }
  }
  __syncthreads();
}

// ---------------------------------------------------------------------------
// conv1: 1x0e -> (16x0e + 8x1e).  w[E,2]; scatter 4 floats/edge.
// ---------------------------------------------------------------------------
__global__ __launch_bounds__(256) void conv1_edge_kernel(
    const float* __restrict__ evec, const int* __restrict__ eidx,
    const float* __restrict__ rW1, const float* __restrict__ rW2,
    const float* __restrict__ sl1, float* __restrict__ acc1)
{
  __shared__ __align__(16) _Float16 sHB[256 * 32];
  __shared__ __align__(16) _Float16 sW1t[64 * 32];
  __shared__ __align__(16) _Float16 sW2t[16 * 64];
  __shared__ __align__(16) _Float16 sH[8 * 16 * 64];
  __shared__ float sOut[256 * 16];
  __shared__ float sN[256 * 3];

  const ColScales cs = {{kMlpO, kMlpO, 0.f, 0.f, 0.f, 0.f}};
  const int eBase = blockIdx.x * 256;
  radial_stage<256, 16>(eBase, evec, rW1, rW2, 2, cs, sHB, sW1t, sW2t, sH, sOut, sN);

  const int tid = threadIdx.x;
  const int e = eBase + tid;
  const int src = eidx[e], dst = eidx[kE + e];
  const float s = sl1[src];
  const float w0 = sOut[tid * 16 + 0];
  const float w1 = sOut[tid * 16 + 1];
  const float yx = kSQ3 * sN[tid * 3 + 0];
  const float yy = kSQ3 * sN[tid * 3 + 1];
  const float yz = kSQ3 * sN[tid * 3 + 2];
  float* acc = acc1 + dst * 4;
  atomicAdd(acc + 0, w0 * s);
  const float c = w1 * s;
  atomicAdd(acc + 1, c * yx);
  atomicAdd(acc + 2, c * yy);
  atomicAdd(acc + 3, c * yz);
}

// ---------------------------------------------------------------------------
// conv2: (8x0e+8x1e) -> (16x0e + 32x1e mid).  w[E,6,8]; scatter 112/edge.
// Irrep path constants (1/sqrt3, 1/sqrt2, C2/sqrt(5/3)) folded into W2 cols.
// Stage 3 uses 2 threads per edge (4 channels each).
// ---------------------------------------------------------------------------
__global__ __launch_bounds__(256) void conv2_edge_kernel(
    const float* __restrict__ evec, const int* __restrict__ eidx,
    const float* __restrict__ rW1, const float* __restrict__ rW2,
    const float* __restrict__ sl, const float* __restrict__ vl,
    float* __restrict__ acc2)
{
  __shared__ __align__(16) _Float16 sHB[128 * 32];
  __shared__ __align__(16) _Float16 sW1t[64 * 32];
  __shared__ __align__(16) _Float16 sW2t[48 * 64];
  __shared__ __align__(16) _Float16 sH[8 * 16 * 64];
  __shared__ float sOut[128 * 48];
  __shared__ float sN[128 * 3];

  const ColScales cs = {{kMlpO, kMlpO, kMlpO,
                         kMlpO * kInvSQ3, kMlpO * kInvSQ2, kMlpO * kInvSQ53 * kC2}};
  const int eBase = blockIdx.x * 128;
  radial_stage<128, 48>(eBase, evec, rW1, rW2, 48, cs, sHB, sW1t, sW2t, sH, sOut, sN);

  const int tid = threadIdx.x;
  const int el = tid & 127, part = tid >> 7;
  const int e = eBase + el;
  const int src = eidx[e], dst = eidx[kE + e];
  const float nx = sN[el * 3 + 0], ny = sN[el * 3 + 1], nz = sN[el * 3 + 2];
  const float yx = kSQ3 * nx, yy = kSQ3 * ny, yz = kSQ3 * nz;
  const float* w   = sOut + el * 48;      // [6][8]
  const float* ssp = sl + src * 8;
  const float* vvp = vl + src * 24;
  float* acc = acc2 + dst * 112;          // [0..15]=s, [16..111]=v(u*3+i)

#pragma unroll
  for (int uu = 0; uu < 4; ++uu) {
    const int u = part * 4 + uu;
    const float ss = ssp[u];
    const float vx = vvp[u * 3 + 0], vy = vvp[u * 3 + 1], vz = vvp[u * 3 + 2];
    const float dotv = vx * yx + vy * yy + vz * yz;
    // m_s0 (w[0]), m_s1 (w[3], 1/sqrt3 folded)
    atomicAdd(acc + u,     w[0 * 8 + u] * ss);
    atomicAdd(acc + 8 + u, w[3 * 8 + u] * dotv);
    // m_v0 = w[1]*ss * y1
    const float c0 = w[1 * 8 + u] * ss;
    atomicAdd(acc + 16 + u * 3 + 0, c0 * yx);
    atomicAdd(acc + 16 + u * 3 + 1, c0 * yy);
    atomicAdd(acc + 16 + u * 3 + 2, c0 * yz);
    // m_v1 = w[2]*vv
    const float c1 = w[2 * 8 + u];
    atomicAdd(acc + 40 + u * 3 + 0, c1 * vx);
    atomicAdd(acc + 40 + u * 3 + 1, c1 * vy);
    atomicAdd(acc + 40 + u * 3 + 2, c1 * vz);
    // m_v2 = w[4]*cross(vv,y1)  (1/sqrt2 folded)
    const float c2 = w[4 * 8 + u];
    atomicAdd(acc + 64 + u * 3 + 0, c2 * (vy * yz - vz * yy));
    atomicAdd(acc + 64 + u * 3 + 1, c2 * (vz * yx - vx * yz));
    atomicAdd(acc + 64 + u * 3 + 2, c2 * (vx * yy - vy * yx));
    // m_v3 = w[5]*(n*(n.vv) - vv/3)  (C2/sqrt(5/3) folded)
    const float ndv = nx * vx + ny * vy + nz * vz;
    const float c3 = w[5 * 8 + u];
    atomicAdd(acc + 88 + u * 3 + 0, c3 * (nx * ndv - vx * (1.0f / 3.0f)));
    atomicAdd(acc + 88 + u * 3 + 1, c3 * (ny * ndv - vy * (1.0f / 3.0f)));
    atomicAdd(acc + 88 + u * 3 + 2, c3 * (nz * ndv - vz * (1.0f / 3.0f)));
  }
}

// ---------------------------------------------------------------------------
// conv3: (8x0e+8x1e) -> 16x0e mid.  w[E,2,8]; scatter 16 floats/edge.
// ---------------------------------------------------------------------------
__global__ __launch_bounds__(256) void conv3_edge_kernel(
    const float* __restrict__ evec, const int* __restrict__ eidx,
    const float* __restrict__ rW1, const float* __restrict__ rW2,
    const float* __restrict__ sl, const float* __restrict__ vl,
    float* __restrict__ acc3)
{
  __shared__ __align__(16) _Float16 sHB[256 * 32];
  __shared__ __align__(16) _Float16 sW1t[64 * 32];
  __shared__ __align__(16) _Float16 sW2t[16 * 64];
  __shared__ __align__(16) _Float16 sH[8 * 16 * 64];
  __shared__ float sOut[256 * 16];
  __shared__ float sN[256 * 3];

  const ColScales cs = {{kMlpO, kMlpO * kInvSQ3, 0.f, 0.f, 0.f, 0.f}};
  const int eBase = blockIdx.x * 256;
  radial_stage<256, 16>(eBase, evec, rW1, rW2, 16, cs, sHB, sW1t, sW2t, sH, sOut, sN);

  const int tid = threadIdx.x;
  const int e = eBase + tid;
  const int src = eidx[e], dst = eidx[kE + e];
  const float yx = kSQ3 * sN[tid * 3 + 0];
  const float yy = kSQ3 * sN[tid * 3 + 1];
  const float yz = kSQ3 * sN[tid * 3 + 2];
  const float* w   = sOut + tid * 16;   // [2][8]
  const float* ssp = sl + src * 8;
  const float* vvp = vl + src * 24;
  float* acc = acc3 + dst * 16;
#pragma unroll
  for (int u = 0; u < 8; ++u) {
    atomicAdd(acc + u, w[u] * ssp[u]);
    const float dotv = vvp[u * 3 + 0] * yx + vvp[u * 3 + 1] * yy + vvp[u * 3 + 2] * yz;
    atomicAdd(acc + 8 + u, w[8 + u] * dotv);   // 1/sqrt3 folded into W2 cols 8..15
  }
}

// ---------------------------------------------------------------------------
// Node-level kernels (tiny dense work, 1 thread per node)
// ---------------------------------------------------------------------------
__global__ __launch_bounds__(256) void k_node_init(
    const int* __restrict__ species, const float* __restrict__ emb,
    const float* __restrict__ lin1Ws, float* __restrict__ s0, float* __restrict__ sl1)
{
  const int n = blockIdx.x * 256 + threadIdx.x;
  if (n >= kN) return;
  const float s = emb[species[n]];
  s0[n] = s;
  sl1[n] = s * lin1Ws[0];
}

__global__ __launch_bounds__(256) void k_node1_post(
    const float* __restrict__ s0, const float* __restrict__ acc1,
    const float* __restrict__ scW0, const float* __restrict__ l2Ws,
    const float* __restrict__ l2Wv,
    float* __restrict__ sfeat, float* __restrict__ vfeat)
{
  const int n = blockIdx.x * 256 + threadIdx.x;
  if (n >= kN) return;
  const float s  = s0[n];
  const float as = acc1[n * 4 + 0] * kISN;
  const float a0 = acc1[n * 4 + 1] * kISN;
  const float a1 = acc1[n * 4 + 2] * kISN;
  const float a2 = acc1[n * 4 + 3] * kISN;
  float outs[16];
#pragma unroll
  for (int j = 0; j < 16; ++j) outs[j] = s * scW0[j] + as * l2Ws[j];
#pragma unroll
  for (int u = 0; u < 8; ++u) sfeat[n * 8 + u] = siluf(outs[u]);
#pragma unroll
  for (int v = 0; v < 8; ++v) {
    const float g = sigmf(outs[8 + v]) * l2Wv[v];
    vfeat[n * 24 + v * 3 + 0] = a0 * g;
    vfeat[n * 24 + v * 3 + 1] = a1 * g;
    vfeat[n * 24 + v * 3 + 2] = a2 * g;
  }
}

__global__ __launch_bounds__(256) void k_node_lin1(
    const float* __restrict__ sfeat, const float* __restrict__ vfeat,
    const float* __restrict__ l1Ws, const float* __restrict__ l1Wv,
    float* __restrict__ sl, float* __restrict__ vl)
{
  const int n = blockIdx.x * 256 + threadIdx.x;
  if (n >= kN) return;
  float si[8], vi[24];
#pragma unroll
  for (int k = 0; k < 8; ++k) si[k] = sfeat[n * 8 + k];
#pragma unroll
  for (int k = 0; k < 24; ++k) vi[k] = vfeat[n * 24 + k];
#pragma unroll
  for (int u = 0; u < 8; ++u) {
    float a = 0.0f;
#pragma unroll
    for (int k = 0; k < 8; ++k) a += si[k] * l1Ws[k * 8 + u];
    sl[n * 8 + u] = a * kInv8;
  }
#pragma unroll
  for (int u = 0; u < 8; ++u)
#pragma unroll
    for (int i = 0; i < 3; ++i) {
      float a = 0.0f;
#pragma unroll
      for (int k = 0; k < 8; ++k) a += vi[k * 3 + i] * l1Wv[k * 8 + u];
      vl[n * 24 + u * 3 + i] = a * kInv8;
    }
}

__global__ __launch_bounds__(256) void k_node2_post(
    const float* __restrict__ acc2,
    const float* __restrict__ scWs, const float* __restrict__ scWv,
    const float* __restrict__ l2Ws, const float* __restrict__ l2Wv,
    float* __restrict__ sfeat, float* __restrict__ vfeat)
{
  const int n = blockIdx.x * 256 + threadIdx.x;
  if (n >= kN) return;
  float sold[8], vold[24];
#pragma unroll
  for (int k = 0; k < 8; ++k) sold[k] = sfeat[n * 8 + k];
#pragma unroll
  for (int k = 0; k < 24; ++k) vold[k] = vfeat[n * 24 + k];
  const float* A = acc2 + n * 112;
  float as[16];
#pragma unroll
  for (int k = 0; k < 16; ++k) as[k] = A[k] * kISN;

  float outs[16];
#pragma unroll
  for (int j = 0; j < 16; ++j) {
    float sc = 0.0f, t = 0.0f;
#pragma unroll
    for (int k = 0; k < 8; ++k) sc += sold[k] * scWs[k * 16 + j];
#pragma unroll
    for (int k = 0; k < 16; ++k) t += as[k] * l2Ws[k * 16 + j];
    outs[j] = sc * kInv8 + t * kInv16;
  }
  float outv[24];
#pragma unroll
  for (int v = 0; v < 8; ++v)
#pragma unroll
    for (int i = 0; i < 3; ++i) {
      float sc = 0.0f, t = 0.0f;
#pragma unroll
      for (int k = 0; k < 8; ++k) sc += vold[k * 3 + i] * scWv[k * 8 + v];
#pragma unroll
      for (int u = 0; u < 32; ++u) t += A[16 + u * 3 + i] * l2Wv[u * 8 + v];
      outv[v * 3 + i] = sc * kInv8 + t * (kISN * kInv32);
    }
#pragma unroll
  for (int u = 0; u < 8; ++u) sfeat[n * 8 + u] = siluf(outs[u]);
#pragma unroll
  for (int v = 0; v < 8; ++v) {
    const float g = sigmf(outs[8 + v]);
#pragma unroll
    for (int i = 0; i < 3; ++i) vfeat[n * 24 + v * 3 + i] = outv[v * 3 + i] * g;
  }
}

__global__ __launch_bounds__(256) void k_node3_head(
    const float* __restrict__ acc3, const float* __restrict__ sfeat,
    const float* __restrict__ scWs /*8x64*/, const float* __restrict__ l2W /*16x64*/,
    const float* __restrict__ hW1 /*64x64*/, const float* __restrict__ hW2 /*64x4*/,
    float* __restrict__ out)
{
  const int n = blockIdx.x * 256 + threadIdx.x;
  if (n >= kN) return;
  float s[8], a[16];
#pragma unroll
  for (int k = 0; k < 8; ++k) s[k] = sfeat[n * 8 + k];
#pragma unroll
  for (int k = 0; k < 16; ++k) a[k] = acc3[n * 16 + k] * kISN;

  float h[64];
  for (int j = 0; j < 64; ++j) {
    float sc = 0.0f, t = 0.0f;
#pragma unroll
    for (int k = 0; k < 8; ++k) sc += s[k] * scWs[k * 64 + j];
#pragma unroll
    for (int k = 0; k < 16; ++k) t += a[k] * l2W[k * 64 + j];
    h[j] = sc * kInv8 + t * kInv16;
  }
  float o[4] = {0.0f, 0.0f, 0.0f, 0.0f};
  for (int k = 0; k < 64; ++k) {
    float t = 0.0f;
    for (int j = 0; j < 64; ++j) t += h[j] * hW1[j * 64 + k];
    const float hh = siluf(0.125f * t);
#pragma unroll
    for (int m = 0; m < 4; ++m) o[m] += hh * hW2[k * 4 + m];
  }
#pragma unroll
  for (int m = 0; m < 4; ++m) out[n * 4 + m] = 0.125f * o[m];
}

// ---------------------------------------------------------------------------
// Host launcher
// ---------------------------------------------------------------------------
extern "C" void kernel_launch(void* const* d_in, const int* in_sizes, int n_in,
                              void* d_out, int out_size, void* d_ws, size_t ws_size,
                              hipStream_t stream)
{
  (void)in_sizes; (void)n_in; (void)out_size; (void)ws_size;

  // setup_inputs() recursive dict (insertion) order:
  const float* emb     = (const float*)d_in[0];   // [8,1]
  const float* c1_scW0 = (const float*)d_in[1];   // [1,16]
  const float* c1_l1Ws = (const float*)d_in[2];   // [1,1]
  const float* c1_rW1  = (const float*)d_in[3];   // [16,64]
  const float* c1_rW2  = (const float*)d_in[4];   // [64,2]
  const float* c1_l2Ws = (const float*)d_in[5];   // [1,16]
  const float* c1_l2Wv = (const float*)d_in[6];   // [1,8]
  const float* c2_scWs = (const float*)d_in[7];   // [8,16]
  const float* c2_scWv = (const float*)d_in[8];   // [8,8]
  const float* c2_l1Ws = (const float*)d_in[9];   // [8,8]
  const float* c2_l1Wv = (const float*)d_in[10];  // [8,8]
  const float* c2_rW1  = (const float*)d_in[11];  // [16,64]
  const float* c2_rW2  = (const float*)d_in[12];  // [64,48]
  const float* c2_l2Ws = (const float*)d_in[13];  // [16,16]
  const float* c2_l2Wv = (const float*)d_in[14];  // [32,8]
  const float* c3_scWs = (const float*)d_in[15];  // [8,64]
  const float* c3_l1Ws = (const float*)d_in[16];  // [8,8]
  const float* c3_l1Wv = (const float*)d_in[17];  // [8,8]
  const float* c3_rW1  = (const float*)d_in[18];  // [16,64]
  const float* c3_rW2  = (const float*)d_in[19];  // [64,16]
  const float* c3_l2W  = (const float*)d_in[20];  // [16,64]
  const float* h_W1    = (const float*)d_in[21];  // [64,64]
  const float* h_W2    = (const float*)d_in[22];  // [64,4]
  const int*   species = (const int*)d_in[23];    // [N]
  const int*   eidx    = (const int*)d_in[24];    // [2,E]
  const float* evec    = (const float*)d_in[25];  // [E,3]

  // workspace carve-up (floats); total = 198*N ~ 39.6 MB
  float* W = (float*)d_ws;
  float* s0    = W; W += kN;
  float* sl1   = W; W += kN;
  float* sfeat = W; W += 8 * kN;
  float* vfeat = W; W += 24 * kN;
  float* slb   = W; W += 8 * kN;
  float* vlb   = W; W += 24 * kN;
  float* acc1  = W; W += 4 * kN;
  float* acc2  = W; W += 112 * kN;
  float* acc3  = W; W += 16 * kN;

  hipMemsetAsync(acc1, 0, sizeof(float) * 4 * kN, stream);
  hipMemsetAsync(acc2, 0, sizeof(float) * 112 * kN, stream);
  hipMemsetAsync(acc3, 0, sizeof(float) * 16 * kN, stream);

  const int NB = (kN + 255) / 256;

  k_node_init<<<NB, 256, 0, stream>>>(species, emb, c1_l1Ws, s0, sl1);

  conv1_edge_kernel<<<kE / 256, 256, 0, stream>>>(evec, eidx, c1_rW1, c1_rW2, sl1, acc1);
  k_node1_post<<<NB, 256, 0, stream>>>(s0, acc1, c1_scW0, c1_l2Ws, c1_l2Wv, sfeat, vfeat);

  k_node_lin1<<<NB, 256, 0, stream>>>(sfeat, vfeat, c2_l1Ws, c2_l1Wv, slb, vlb);
  conv2_edge_kernel<<<kE / 128, 256, 0, stream>>>(evec, eidx, c2_rW1, c2_rW2, slb, vlb, acc2);
  k_node2_post<<<NB, 256, 0, stream>>>(acc2, c2_scWs, c2_scWv, c2_l2Ws, c2_l2Wv, sfeat, vfeat);

  k_node_lin1<<<NB, 256, 0, stream>>>(sfeat, vfeat, c3_l1Ws, c3_l1Wv, slb, vlb);
  conv3_edge_kernel<<<kE / 256, 256, 0, stream>>>(evec, eidx, c3_rW1, c3_rW2, slb, vlb, acc3);
  k_node3_head<<<NB, 256, 0, stream>>>(acc3, sfeat, c3_scWs, c3_l2W, h_W1, h_W2, (float*)d_out);
}
